// TwoDimensionalSSM_10917806866569
// MI455X (gfx1250) — matrix operand order repaired and checked by reading, not compiled, not run on hardware
//
#include <hip/hip_runtime.h>
#include <hip/hip_bf16.h>
#include <math.h>

// 2D-SSM: per-channel impulse kernel + depthwise causal 2D conv via WMMA,
// fused residual + SiLU.  Target: gfx1250 (wave32, v_wmma_f32_16x16x32_f16).

typedef __attribute__((ext_vector_type(16))) _Float16 v16h;
typedef __attribute__((ext_vector_type(8)))  float    v8f;

#define SEQ   1024
#define BSZ   16
#define DIM   1024
#define NSIDE 32
#define SCALE 0.70710678118654752f   // sqrt(1/2)

__device__ __forceinline__ float sigmoidf_(float x) {
    return 1.0f / (1.0f + __expf(-x));
}

// ---------------------------------------------------------------------------
// Kernel 1: build impulse-response kernel K[d][p][q]  (D x 32 x 32, f32)
// One thread per channel d; per-column (h,v) state lives in LDS.
// Recurrence (impulse at (0,0)):
//   v[i,j] = a2*h[i-1,j] + a1*v[i-1,j] + b2*x[i,j]
//   h[i,j] = a1*h[i,j-1] + a2*v[i,j-1] + b1*x[i,j]
//   K[i,j] = <h,C1*s> + <v,C2*s>
// ---------------------------------------------------------------------------
__global__ __launch_bounds__(64) void ssm_build_K(
    const float* __restrict__ A1, const float* __restrict__ A2,
    const float* __restrict__ B1, const float* __restrict__ B2,
    const float* __restrict__ C1, const float* __restrict__ C2,
    float* __restrict__ Kw)
{
    __shared__ float sh[NSIDE][2][64];   // h state per column, per thread
    __shared__ float sv[NSIDE][2][64];   // v state per column, per thread

    const int t = threadIdx.x;
    const int d = blockIdx.x * 64 + t;

    float a1[2], a2[2], b1[2], b2[2], c1s[2], c2s[2];
#pragma unroll
    for (int c = 0; c < 2; ++c) {
        a1[c]  = sigmoidf_(A1[d * 2 + c]) * 0.5f;
        a2[c]  = sigmoidf_(A2[d * 2 + c]) * 0.5f;
        b1[c]  = sigmoidf_(B1[d * 2 + c]) * 0.5f;
        b2[c]  = sigmoidf_(B2[d * 2 + c]) * 0.5f;
        c1s[c] = C1[d * 2 + c] * SCALE;
        c2s[c] = C2[d * 2 + c] * SCALE;
    }

    for (int j = 0; j < NSIDE; ++j) {
        sh[j][0][t] = 0.0f; sh[j][1][t] = 0.0f;
        sv[j][0][t] = 0.0f; sv[j][1][t] = 0.0f;
    }

    float* Ko = Kw + (size_t)d * (NSIDE * NSIDE);

    for (int i = 0; i < NSIDE; ++i) {
        // v_row = a2*h_prev + a1*v_prev (+ b2 at impulse)
        for (int j = 0; j < NSIDE; ++j) {
#pragma unroll
            for (int c = 0; c < 2; ++c) {
                float nv = a2[c] * sh[j][c][t] + a1[c] * sv[j][c][t];
                if (i == 0 && j == 0) nv += b2[c];
                sv[j][c][t] = nv;
            }
        }
        // h scan across columns; emit K
        float hl0 = 0.0f, hl1 = 0.0f;
        for (int j = 0; j < NSIDE; ++j) {
            float s0 = (j > 0) ? sv[j - 1][0][t] : 0.0f;
            float s1 = (j > 0) ? sv[j - 1][1][t] : 0.0f;
            float h0 = a1[0] * hl0 + a2[0] * s0;
            float h1 = a1[1] * hl1 + a2[1] * s1;
            if (i == 0 && j == 0) { h0 += b1[0]; h1 += b1[1]; }
            sh[j][0][t] = h0; sh[j][1][t] = h1;
            hl0 = h0; hl1 = h1;
            Ko[i * NSIDE + j] = h0 * c1s[0] + h1 * c1s[1]
                              + sv[j][0][t] * c2s[0] + sv[j][1][t] * c2s[1];
        }
    }
}

// ---------------------------------------------------------------------------
// Kernel 2: depthwise causal conv as block-Toeplitz WMMA GEMMs.
// One block (8 waves) per channel d.
//   Out_i(32j x 16b) = sum_{p=0..i} L_p(32x32) @ Xrow_{i-p}(32x16)
//   L_p[j,j'] = K[d][p][j-j'] (lower-tri Toeplitz), 2 M-tiles of 16.
// A-tiles double-buffered over p in two 16-p phases -> 64KB static LDS.
// ---------------------------------------------------------------------------
__global__ __launch_bounds__(256) void ssm_conv_wmma(
    const float* __restrict__ xg, const float* __restrict__ Kw,
    const float* __restrict__ omega, float* __restrict__ out)
{
    // B operands: xs[r][lane][h] = x[row r, col (h + 16*(lane>>4)), b=lane&15]
    __shared__ __align__(32) _Float16 xs[NSIDE][32][16];      // 32 KB
    // A operands for 16 p's: as_[p'][mt][lane][h]
    __shared__ __align__(32) _Float16 as_[16][2][32][16];     // 32 KB

    const int d    = blockIdx.x;
    const int tid  = threadIdx.x;
    const int lane = tid & 31;
    const int w    = tid >> 5;

    // ---- stage x[:, :, d] into WMMA-B layout (f16) ----
    for (int e = tid; e < SEQ * BSZ; e += 256) {
        const int b = e & 15;
        const int s = e >> 4;
        const int r = s >> 5;            // image row
        const int c = s & 31;            // image col = contraction index k
        const int L = ((c >> 4) << 4) | b;
        const int h = c & 15;
        xs[r][L][h] = (_Float16)xg[(size_t)s * (BSZ * DIM) + b * DIM + d];
    }

    const float* Kd = Kw + (size_t)d * (NSIDE * NSIDE);
    const float  om = omega[d];

    v8f acc[4][2];
#pragma unroll
    for (int ii = 0; ii < 4; ++ii) { acc[ii][0] = (v8f){}; acc[ii][1] = (v8f){}; }

    for (int phase = 0; phase < 2; ++phase) {
        const int pbase = phase * 16;
        __syncthreads();   // phase0: xs ready; phase1: prior readers of as_ done

        // Build A tiles for p in [pbase, pbase+16), contiguous k order (same as xs):
        // lane L holds row (L&15); half h -> K = h + 16*(L>>4)
        for (int e = tid; e < 16 * 2 * 32 * 16; e += 256) {
            const int h  = e & 15;
            const int L  = (e >> 4) & 31;
            const int mt = (e >> 9) & 1;
            const int pp = e >> 10;
            const int hi = L >> 4;
            const int m  = mt * 16 + (L & 15);
            const int k  = h + 16 * hi;
            const float v = (k <= m) ? Kd[(pbase + pp) * NSIDE + (m - k)] : 0.0f;
            ((_Float16*)as_)[e] = (_Float16)v;
        }
        __syncthreads();

#pragma unroll
        for (int ii = 0; ii < 4; ++ii) {
            const int i = ii * 8 + w;                 // output image row
            const int pend = (i < pbase + 15) ? i : (pbase + 15);
            for (int p = pbase; p <= pend; ++p) {
                v16h B  = *(const v16h*)(&xs[i - p][lane][0]);
                v16h A0 = *(const v16h*)(&as_[p - pbase][0][lane][0]);
                v16h A1 = *(const v16h*)(&as_[p - pbase][1][lane][0]);
                acc[ii][0] = __builtin_amdgcn_wmma_f32_16x16x32_f16(
                    false, A0, false, B, (short)0, acc[ii][0], false, false);
                acc[ii][1] = __builtin_amdgcn_wmma_f32_16x16x32_f16(
                    false, A1, false, B, (short)0, acc[ii][1], false, false);
            }
        }
    }

    // ---- fused residual + SiLU + store ----
    // C/D layout: VGPR r, lane L -> (row = r + 8*(L>>4) [+16*mt], col = L&15)
    const int n  = lane & 15;
    const int hi = lane >> 4;
#pragma unroll
    for (int ii = 0; ii < 4; ++ii) {
        const int i = ii * 8 + w;
#pragma unroll
        for (int mt = 0; mt < 2; ++mt) {
#pragma unroll
            for (int r = 0; r < 8; ++r) {
                const int j = mt * 16 + hi * 8 + r;
                const size_t g = (size_t)(i * NSIDE + j) * (BSZ * DIM) + n * DIM + d;
                const float val = acc[ii][mt][r] + xg[g] * om;
                out[g] = val * (1.0f / (1.0f + __expf(-val)));
            }
        }
    }
}

// ---------------------------------------------------------------------------
extern "C" void kernel_launch(void* const* d_in, const int* in_sizes, int n_in,
                              void* d_out, int out_size, void* d_ws, size_t ws_size,
                              hipStream_t stream) {
    (void)in_sizes; (void)n_in; (void)out_size; (void)ws_size;
    const float* x     = (const float*)d_in[0];
    const float* A1    = (const float*)d_in[1];
    const float* A2    = (const float*)d_in[2];
    const float* B1    = (const float*)d_in[3];
    const float* B2    = (const float*)d_in[4];
    const float* C1    = (const float*)d_in[5];
    const float* C2    = (const float*)d_in[6];
    const float* omega = (const float*)d_in[7];

    float* Kw = (float*)d_ws;            // D*32*32 f32 = 4 MB scratch

    ssm_build_K<<<DIM / 64, 64, 0, stream>>>(A1, A2, B1, B2, C1, C2, Kw);
    ssm_conv_wmma<<<DIM, 256, 0, stream>>>(x, Kw, omega, (float*)d_out);
}
